// attention_44324062495123
// MI455X (gfx1250) — matrix operand order repaired and checked by reading, not compiled, not run on hardware
//
#include <hip/hip_runtime.h>
#include <hip/hip_fp16.h>

// ---------------------------------------------------------------------------
// MI455X (gfx1250, wave32) fused multi-head attention via v_wmma_f32_16x16x32_f16
//   out[b,n,:] = sum_m softmax( (H Wq_m)(H Wk_m)^T / 8 ) (H Wv_m) Wo_m
// Pipeline: prep (fold scale*log2e, f16 repack) -> QKV GEMM -> flash attn -> out GEMM
// All LDS staging is double-buffered (one barrier per K-step).
// ---------------------------------------------------------------------------

typedef __attribute__((ext_vector_type(16))) _Float16 v16h;
typedef __attribute__((ext_vector_type(8)))  _Float16 h8;
typedef __attribute__((ext_vector_type(8)))  float    v8f;

constexpr int Mh   = 8;     // heads
constexpr int DH   = 512;   // model dim
constexpr int DKV  = 64;    // head dim (k == v)
constexpr int Bc   = 16;    // batch
constexpr int Nc   = 1024;  // sequence
constexpr int QKVC = 3 * Mh * DKV; // 1536 fused projection columns

// 1/sqrt(64) * log2(e): softmax done in exp2 domain (constant cancels in the
// normalization, running max is monotone-invariant).
#define QSCALE_LOG2E 0.1803368801111204f

// ---- fragment helpers (layouts per CDNA5 ISA §7.12.2) ----------
// A 16x32 f16: lane l holds row (l&15); halves 0..7 = K base8+0..7,
// halves 8..15 = K 16+base8+0..7, base8 = (l>=16)?8:0.  Two 16B loads.
static __device__ __forceinline__ v16h load_a_frag(const _Float16* p, int base8) {
  h8 lo = *(const h8*)(p + base8);
  h8 hi = *(const h8*)(p + 16 + base8);
  v16h a;
#pragma unroll
  for (int i = 0; i < 8; ++i) { a[i] = lo[i]; a[i + 8] = hi[i]; }
  return a;
}
// B 32x16 f16: lane l holds col (l&15); halves 0..7 = K base8+0..7,
// halves 8..15 = K 16+base8+0..7, base8 = base16/2 = (l>=16)?8:0.
// p points to col start (k-contiguous storage).
static __device__ __forceinline__ v16h load_b_frag(const _Float16* p, int base16) {
  h8 lo = *(const h8*)(p + (base16 >> 1));
  h8 hi = *(const h8*)(p + 16 + (base16 >> 1));
  v16h b;
#pragma unroll
  for (int i = 0; i < 8; ++i) { b[i] = lo[i]; b[i + 8] = hi[i]; }
  return b;
}
static __device__ __forceinline__ v8f wmma_f16(v16h a, v16h b, v8f c) {
  return __builtin_amdgcn_wmma_f32_16x16x32_f16(false, a, false, b, (short)0, c,
                                                false, false);
}

// ---------------------------------------------------------------------------
// Kernel 1: weight repack to f16, col-major (K-contiguous) for B-fragments.
//  Wqkv_t[c][d], c = {Q: m*64+k | K: 512+.. | V: 1024+..}; Q scaled by
//  0.125*log2e.  Wo_t[dout][r], r = m*64+v.
// ---------------------------------------------------------------------------
__global__ __launch_bounds__(256) void prep_weights(
    const float* __restrict__ Wq, const float* __restrict__ Wk,
    const float* __restrict__ Wv, const float* __restrict__ Wo,
    _Float16* __restrict__ Wqkv_t, _Float16* __restrict__ Wo_t) {
  int i = blockIdx.x * 256 + threadIdx.x;
  const int n1 = QKVC * DH;
  const int n2 = DH * DH;
  if (i < n1) {
    int c = i >> 9, d = i & 511;
    float v;
    if (c < 512)       { int m = c >> 6, k = c & 63;
                         v = Wq[((size_t)m * DH + d) * DKV + k] * QSCALE_LOG2E; }
    else if (c < 1024) { int c2 = c - 512; int m = c2 >> 6, k = c2 & 63;
                         v = Wk[((size_t)m * DH + d) * DKV + k]; }
    else               { int c2 = c - 1024; int m = c2 >> 6, k = c2 & 63;
                         v = Wv[((size_t)m * DH + d) * DKV + k]; }
    Wqkv_t[(size_t)c * DH + d] = (_Float16)v;
  } else if (i < n1 + n2) {
    int j = i - n1;
    int dout = j >> 9, r = j & 511;
    int m = r >> 6, v = r & 63;
    Wo_t[(size_t)dout * DH + r] = (_Float16)Wo[((size_t)(m * DKV) + v) * DH + dout];
  }
}

// ---------------------------------------------------------------------------
// Kernels 2 & 4: tiled WMMA GEMM, C[b] = A[b](1024 x 512) * Bt^T.
//  Bt is col-major (K-contiguous, stride 512).  BM=BN=128, BK=32, 8 waves,
//  each wave owns a 32x64 tile = 2x4 WMMA frags.  Double-buffered LDS; rows
//  padded to 40 halves (80B: 16B-aligned, conflict-free fragment reads).
// ---------------------------------------------------------------------------
template <bool A_F32, bool OUT_F16>
__global__ __launch_bounds__(256) void gemm_wmma(
    const void* __restrict__ Asrc, const _Float16* __restrict__ Bt,
    void* __restrict__ Out, int Ncol) {
  constexpr int BK = 32, STR = 40, KT = DH / BK;
  __shared__ _Float16 As[2][128 * STR];
  __shared__ _Float16 Bs[2][128 * STR];
  const int b = blockIdx.z;
  const int rowBase = blockIdx.y * 128;
  const int colBase = blockIdx.x * 128;
  const int tid = threadIdx.x, lane = tid & 31, wid = tid >> 5;
  const int wr = wid & 3, wc = wid >> 2;      // 4x2 wave grid
  const int l15 = lane & 15;
  const int base8  = (lane & 16) ? 8 : 0;
  const int base16 = (lane & 16) ? 16 : 0;
  const int hi8    = (lane & 16) ? 8 : 0;

  const float*    Af = (const float*)Asrc    + (size_t)b * Nc * DH;
  const _Float16* Ah = (const _Float16*)Asrc + (size_t)b * Nc * DH;

  auto stage = [&](int kt, int sel) {
    if constexpr (A_F32) {
#pragma unroll
      for (int i = 0; i < 4; ++i) {           // 128 rows x 8 float4 segs
        int idx = tid + i * 256;
        int row = idx >> 3, seg = idx & 7;
        float4 f = *(const float4*)(Af + (size_t)(rowBase + row) * DH +
                                    kt * BK + seg * 4);
        _Float16* d = &As[sel][row * STR + seg * 4];
        d[0] = (_Float16)f.x; d[1] = (_Float16)f.y;
        d[2] = (_Float16)f.z; d[3] = (_Float16)f.w;
      }
    } else {
#pragma unroll
      for (int i = 0; i < 2; ++i) {           // 128 rows x 4 h8 segs
        int idx = tid + i * 256;
        int row = idx >> 2, seg = idx & 3;
        h8 v = *(const h8*)(Ah + (size_t)(rowBase + row) * DH + kt * BK + seg * 8);
        *(h8*)&As[sel][row * STR + seg * 8] = v;
      }
    }
#pragma unroll
    for (int i = 0; i < 2; ++i) {             // 128 cols x 4 h8 segs
      int idx = tid + i * 256;
      int col = idx >> 2, seg = idx & 3;
      h8 v = *(const h8*)(Bt + (size_t)(colBase + col) * DH + kt * BK + seg * 8);
      *(h8*)&Bs[sel][col * STR + seg * 8] = v;
    }
  };

  v8f acc[2][4];
#pragma unroll
  for (int i = 0; i < 2; ++i)
#pragma unroll
    for (int j = 0; j < 4; ++j) acc[i][j] = {};

  stage(0, 0);
  for (int kt = 0; kt < KT; ++kt) {
    __syncthreads();                          // staged tile `kt` visible
    const int cur = kt & 1;
    if (kt + 1 < KT) stage(kt + 1, cur ^ 1);  // overlap next slab with WMMA

    v16h aF[2], bF[4];
#pragma unroll
    for (int ti = 0; ti < 2; ++ti)
      aF[ti] = load_a_frag(&As[cur][(wr * 32 + ti * 16 + l15) * STR], base8);
#pragma unroll
    for (int tj = 0; tj < 4; ++tj)
      bF[tj] = load_b_frag(&Bs[cur][(wc * 64 + tj * 16 + l15) * STR], base16);
#pragma unroll
    for (int ti = 0; ti < 2; ++ti)
#pragma unroll
      for (int tj = 0; tj < 4; ++tj)
        acc[ti][tj] = wmma_f16(aF[ti], bF[tj], acc[ti][tj]);
  }

  // C layout: lane (l&15) = col, VGPR r = row r + hi8.
  if (OUT_F16) {
    _Float16* O = (_Float16*)Out + (size_t)b * Nc * Ncol;
#pragma unroll
    for (int ti = 0; ti < 2; ++ti)
#pragma unroll
      for (int tj = 0; tj < 4; ++tj)
#pragma unroll
        for (int r = 0; r < 8; ++r) {
          int row = rowBase + wr * 32 + ti * 16 + r + hi8;
          int col = colBase + wc * 64 + tj * 16 + l15;
          O[(size_t)row * Ncol + col] = (_Float16)acc[ti][tj][r];
        }
  } else {
    float* O = (float*)Out + (size_t)b * Nc * Ncol;
#pragma unroll
    for (int ti = 0; ti < 2; ++ti)
#pragma unroll
      for (int tj = 0; tj < 4; ++tj)
#pragma unroll
        for (int r = 0; r < 8; ++r) {
          int row = rowBase + wr * 32 + ti * 16 + r + hi8;
          int col = colBase + wc * 64 + tj * 16 + l15;
          O[(size_t)row * Ncol + col] = acc[ti][tj][r];
        }
  }
}

// ---------------------------------------------------------------------------
// Kernel 3: flash attention per (b, head, 64-query block).  4 waves, each
// owns 16 query rows.  Q A-frags straight from global; K staged row-major;
// V staged transposed; both double-buffered.  Softmax in exp2 domain
// (log2e folded into Wq).  P bounced through per-wave LDS for C->A relayout.
// ---------------------------------------------------------------------------
__global__ __launch_bounds__(128) void flash_attn(
    const _Float16* __restrict__ QKV, _Float16* __restrict__ Hp) {
  constexpr int STR = 72;  // 144B row stride: 16B aligned, conflict-free
  constexpr int NKB = Nc / 64;
  __shared__ _Float16 Klds[2][64 * STR];
  __shared__ _Float16 Vlds[2][64 * STR];    // transposed: [v][key]
  __shared__ _Float16 Plds[4 * 16 * STR];   // per-wave P tile
  int bid = blockIdx.x;
  const int qb = bid & 15;  bid >>= 4;      // query block (N/64)
  const int m  = bid & 7;   bid >>= 3;      // head
  const int b  = bid;                       // batch
  const int tid = threadIdx.x, lane = tid & 31, wave = tid >> 5;
  const int l15 = lane & 15;
  const int base8  = (lane & 16) ? 8 : 0;
  const int base16 = (lane & 16) ? 16 : 0;
  const int hi8    = (lane & 16) ? 8 : 0;
  const size_t rs = QKVC;

  // Q fragments (scale*log2e already folded): row-major global, k-contiguous
  const _Float16* Qb =
      QKV + ((size_t)b * Nc + qb * 64 + wave * 16 + l15) * rs + m * 64;
  v16h aQ[2];
  aQ[0] = load_a_frag(Qb + 0,  base8);
  aQ[1] = load_a_frag(Qb + 32, base8);

  auto stageKV = [&](int kb, int sel) {
    const _Float16* Kb = QKV + ((size_t)b * Nc + kb * 64) * rs + 512  + m * 64;
    const _Float16* Vb = QKV + ((size_t)b * Nc + kb * 64) * rs + 1024 + m * 64;
#pragma unroll
    for (int i = 0; i < 4; ++i) {            // 64 rows x 8 h8 segs, 128 thr
      int idx = tid + i * 128;
      int row = idx >> 3, seg = idx & 7;
      h8 kv = *(const h8*)(Kb + (size_t)row * rs + seg * 8);
      *(h8*)&Klds[sel][row * STR + seg * 8] = kv;
      h8 vv = *(const h8*)(Vb + (size_t)row * rs + seg * 8);
#pragma unroll
      for (int j = 0; j < 8; ++j) Vlds[sel][(seg * 8 + j) * STR + row] = vv[j];
    }
  };

  v8f Oa[4];
#pragma unroll
  for (int t = 0; t < 4; ++t) Oa[t] = {};
  float m_run[8], l_run[8];
#pragma unroll
  for (int r = 0; r < 8; ++r) { m_run[r] = -3.0e38f; l_run[r] = 0.f; }

  _Float16* Pw = &Plds[wave * 16 * STR];

  stageKV(0, 0);
  for (int kb = 0; kb < NKB; ++kb) {
    __syncthreads();                          // tile `kb` visible
    const int cur = kb & 1;
    if (kb + 1 < NKB) stageKV(kb + 1, cur ^ 1);  // overlap with compute

    // S = Q K^T : 4 key tiles x 2 k-chunks (S already in log2 units)
    v8f s[4];
#pragma unroll
    for (int t = 0; t < 4; ++t) {
      s[t] = {};
#pragma unroll
      for (int kc = 0; kc < 2; ++kc) {
        v16h bK = load_b_frag(&Klds[cur][(t * 16 + l15) * STR + kc * 32], base16);
        s[t] = wmma_f16(aQ[kc], bK, s[t]);
      }
    }

    // online softmax: rows 0-7 in lanes 0-15, rows 8-15 in lanes 16-31
    float rmax[8];
#pragma unroll
    for (int r = 0; r < 8; ++r)
      rmax[r] = fmaxf(fmaxf(s[0][r], s[1][r]), fmaxf(s[2][r], s[3][r]));
#pragma unroll
    for (int mask = 1; mask < 16; mask <<= 1)
#pragma unroll
      for (int r = 0; r < 8; ++r)
        rmax[r] = fmaxf(rmax[r], __shfl_xor(rmax[r], mask, 16));
    float alpha[8], rsum[8];
#pragma unroll
    for (int r = 0; r < 8; ++r) {
      float mn = fmaxf(m_run[r], rmax[r]);
      alpha[r] = exp2f(m_run[r] - mn);
      m_run[r] = mn;
      rsum[r] = 0.f;
    }
#pragma unroll
    for (int t = 0; t < 4; ++t)
#pragma unroll
      for (int r = 0; r < 8; ++r) {
        float pv = exp2f(s[t][r] - m_run[r]);
        rsum[r] += pv;
        Pw[(r + hi8) * STR + t * 16 + l15] = (_Float16)pv;  // C->A via LDS
      }
#pragma unroll
    for (int mask = 1; mask < 16; mask <<= 1)
#pragma unroll
      for (int r = 0; r < 8; ++r) rsum[r] += __shfl_xor(rsum[r], mask, 16);
#pragma unroll
    for (int r = 0; r < 8; ++r) l_run[r] = alpha[r] * l_run[r] + rsum[r];
#pragma unroll
    for (int t = 0; t < 4; ++t)
#pragma unroll
      for (int r = 0; r < 8; ++r) Oa[t][r] *= alpha[r];

    // O += P V  (per-wave P tile; same-wave DS ops are in-order)
#pragma unroll
    for (int kc = 0; kc < 2; ++kc) {
      v16h aP = load_a_frag(&Pw[l15 * STR + kc * 32], base8);
#pragma unroll
      for (int t = 0; t < 4; ++t) {
        v16h bV = load_b_frag(&Vlds[cur][(t * 16 + l15) * STR + kc * 32], base16);
        Oa[t] = wmma_f16(aP, bV, Oa[t]);
      }
    }
  }

  // epilogue: normalize, store head context (head-major columns)
  _Float16* Ob = Hp + ((size_t)b * Nc + qb * 64 + wave * 16) * 512 + m * 64;
#pragma unroll
  for (int r = 0; r < 8; ++r) {
    float inv = 1.f / l_run[r];
#pragma unroll
    for (int t = 0; t < 4; ++t)
      Ob[(size_t)(r + hi8) * 512 + t * 16 + l15] = (_Float16)(Oa[t][r] * inv);
  }
}

// ---------------------------------------------------------------------------
extern "C" void kernel_launch(void* const* d_in, const int* in_sizes, int n_in,
                              void* d_out, int out_size, void* d_ws, size_t ws_size,
                              hipStream_t stream) {
  const float* H  = (const float*)d_in[0];
  const float* Wq = (const float*)d_in[1];
  const float* Wk = (const float*)d_in[2];
  const float* Wv = (const float*)d_in[3];
  const float* Wo = (const float*)d_in[4];
  float* out = (float*)d_out;

  char* ws = (char*)d_ws;
  _Float16* Wqkv_t = (_Float16*)(ws);                 // 1536*512*2 = 1.5 MB
  _Float16* Wo_t   = (_Float16*)(ws + (1536u * 512u * 2u));
  _Float16* QKV    = (_Float16*)(ws + 2097152u);      // 16*1024*1536*2 = 48 MB
  _Float16* Hp     = (_Float16*)(ws + 52428800u);     // 16*1024*512*2  = 16 MB
  (void)in_sizes; (void)n_in; (void)out_size; (void)ws_size;

  prep_weights<<<dim3((1536 * 512 + 512 * 512) / 256), 256, 0, stream>>>(
      Wq, Wk, Wv, Wo, Wqkv_t, Wo_t);
  gemm_wmma<true, true><<<dim3(QKVC / 128, Nc / 128, Bc), 256, 0, stream>>>(
      H, Wqkv_t, QKV, QKVC);
  flash_attn<<<dim3(Bc * Mh * (Nc / 64)), 128, 0, stream>>>(QKV, Hp);
  gemm_wmma<false, false><<<dim3(DH / 128, Nc / 128, Bc), 256, 0, stream>>>(
      Hp, Wo_t, out, DH);
}
